// MultiHashCodebookKIFLayer_54039278518744
// MI455X (gfx1250) — matrix operand order repaired and checked by reading, not compiled, not run on hardware
//
#include <hip/hip_runtime.h>

// ---------------------------------------------------------------------------
// MI455X / gfx1250, wave32. Static problem config from the reference:
//   B=4096, F=24, S=6, I=123, H=2, E=16 (H*E=32), EO=16, D=16, NB=2e6
// ---------------------------------------------------------------------------

typedef __attribute__((ext_vector_type(16))) __bf16 v16bf;
typedef __attribute__((ext_vector_type(8)))  __bf16 v8bf;
typedef __attribute__((ext_vector_type(4)))  __bf16 v4bf;
typedef __attribute__((ext_vector_type(8)))  float  v8f;

#define BATCH 4096
#define F_    24
#define I_    123
#define D_    16
#define ZDIM  384   // F*EO
#define NPAD2 128   // I_ padded for GEMM2 B operand / LDS tables
#define ACCSZ (F_ * D_ + D_)   // field accumulators + 16-float dummy pad slot

static __device__ __forceinline__ __bf16 tobf(float f) { return (__bf16)f; }

union ABFrag { v16bf v; v8bf h[2]; };

// ---------------------------------------------------------------------------
// Prep kernels: one-shot fp32 -> bf16 conversions, hoisted out of hot loops.
// ---------------------------------------------------------------------------
__global__ __launch_bounds__(256)
void conv_bf16(const float* __restrict__ src, __bf16* __restrict__ dst, int n4) {
  const int t = blockIdx.x * blockDim.x + threadIdx.x;
  if (t >= n4) return;
  const float4 v = reinterpret_cast<const float4*>(src)[t];
  v4bf o; o[0] = tobf(v.x); o[1] = tobf(v.y); o[2] = tobf(v.z); o[3] = tobf(v.w);
  reinterpret_cast<v4bf*>(dst)[t] = o;            // single 8B store
}

// src [K,N] fp32 row-major -> dst [Npad,K] bf16 (transposed, zero-padded rows)
__global__ __launch_bounds__(256)
void transpose_bf16(const float* __restrict__ src, __bf16* __restrict__ dst,
                    int K, int N, int Npad) {
  const int t = blockIdx.x * blockDim.x + threadIdx.x;
  if (t >= Npad * K) return;
  const int n = t / K, k = t % K;
  dst[t] = tobf((n < N) ? src[(size_t)k * N + n] : 0.0f);
}

// ---------------------------------------------------------------------------
// bf16 GEMM, one 16(M) x 64(N) tile per wave: per k-step one A fragment feeds
// 4 WMMAs (4 accumulators), quadrupling A reuse.
//   A  : [M,K]    bf16 row-major
//   BT : [Npad,K] bf16 (B transposed, zero-padded) -> per-lane loads contiguous
//   C  : fp32 [M,N] (store-guarded) or bf16 [M,N]
// nG = Npad/64 column groups.
// ---------------------------------------------------------------------------
template <bool OUT_BF16>
__global__ __launch_bounds__(256)
void gemm_wmma_bf16(const __bf16* __restrict__ A,
                    const __bf16* __restrict__ BT,
                    void* __restrict__ Cout,
                    int M, int N, int K, int nG) {
  const int wid  = threadIdx.x >> 5;
  const int lane = threadIdx.x & 31;
  const int tile = blockIdx.x * (blockDim.x >> 5) + wid;
  if (tile >= (M >> 4) * nG) return;             // wave-uniform

  const int tm   = tile / nG;
  const int tg   = tile % nG;                    // 64-wide column group
  const int half = lane >> 4;                    // 0 | 1
  const int sub  = lane & 15;
  const int m    = (tm << 4) + sub;              // A row for this lane
  const int n0   = (tg << 6) + sub;              // first of 4 columns

  const __bf16* arow = A + (size_t)m * K;
  const __bf16* brow = BT + (size_t)n0 * K;      // zero-padded: always valid
  const size_t  bcs  = (size_t)16 * K;           // +16 columns

  v8f acc0 = {}, acc1 = {}, acc2 = {}, acc3 = {};
  for (int k0 = 0; k0 < K; k0 += 32) {
    ABFrag a;                                    // elems 0..15: K[k0+half*16..]
    a.h[0] = *reinterpret_cast<const v8bf*>(arow + k0 + half * 16);
    a.h[1] = *reinterpret_cast<const v8bf*>(arow + k0 + half * 16 + 8);
    const int kb = k0 + half * 16;               // B: lanes 0-15 K k0..k0+15
    const v16bf b0 = *reinterpret_cast<const v16bf*>(brow + kb);
    const v16bf b1 = *reinterpret_cast<const v16bf*>(brow + bcs + kb);
    const v16bf b2 = *reinterpret_cast<const v16bf*>(brow + 2 * bcs + kb);
    const v16bf b3 = *reinterpret_cast<const v16bf*>(brow + 3 * bcs + kb);
    acc0 = __builtin_amdgcn_wmma_f32_16x16x32_bf16(false, a.v, false, b0, (short)0, acc0, false, false);
    acc1 = __builtin_amdgcn_wmma_f32_16x16x32_bf16(false, a.v, false, b1, (short)0, acc1, false, false);
    acc2 = __builtin_amdgcn_wmma_f32_16x16x32_bf16(false, a.v, false, b2, (short)0, acc2, false, false);
    acc3 = __builtin_amdgcn_wmma_f32_16x16x32_bf16(false, a.v, false, b3, (short)0, acc3, false, false);
  }

  v8f accs[4] = {acc0, acc1, acc2, acc3};
#pragma unroll
  for (int g = 0; g < 4; ++g) {
    const int n = n0 + g * 16;
    if (OUT_BF16) {
      __bf16* C = reinterpret_cast<__bf16*>(Cout);
#pragma unroll
      for (int v = 0; v < 8; ++v) {
        const int mr = (tm << 4) + v + half * 8;
        C[(size_t)mr * N + n] = tobf(accs[g][v]);
      }
    } else if (n < N) {
      float* C = reinterpret_cast<float*>(Cout);
#pragma unroll
      for (int v = 0; v < 8; ++v) {
        const int mr = (tm << 4) + v + half * 8;
        C[(size_t)mr * N + n] = accs[g][v];
      }
    }
  }
}

// ---------------------------------------------------------------------------
// Fused: codebook gather -> te = merged @ W_t (WMMA) -> scale by w[b,i] ->
// segment-sum over endpoint fields in LDS -> out[b, F, D].
// One block (8 waves) per batch element; wave r owns interactions 16r..16r+15.
// Per-interaction metadata (excitation weight + packed field offsets) is
// built once per block into LDS; the scatter loop is branch-free: padded rows
// carry weight 0 and their offsets point at a dummy LDS slot past the real
// accumulators, so every element does ds_load_b64 + 2x ds_add_f32, no cmpx.
// ---------------------------------------------------------------------------
__global__ __launch_bounds__(256)
void fused_te_senetsum(const int*    __restrict__ hash_idx,  // [B, I, 2]
                       const float*  __restrict__ codebook,  // [NB, 16]
                       const __bf16* __restrict__ WtT,       // [16, 32] bf16 (W_t^T)
                       const float*  __restrict__ w,         // [B, I]
                       float*        __restrict__ out) {     // [B, F, 16]
  __shared__ float  sacc[ACCSZ];                 // [0,384) fields, [384,400) dummy
  __shared__ float2 sdat[NPAD2];                 // {w[b,i], packed fi/fj offsets}
  const int b = blockIdx.x;
  for (int t = threadIdx.x; t < ACCSZ; t += blockDim.x) sacc[t] = 0.0f;
  for (int t = threadIdx.x; t < NPAD2; t += blockDim.x) {
    const float wv = (t < I_) ? w[(size_t)b * I_ + t] : 0.0f;
    // branchless pair decode: cum(i) = 23i - i(i-1)/2; pairs (i<6, j>i)
    const int fi = (t >= 23) + (t >= 45) + (t >= 66) + (t >= 86) + (t >= 105);
    const int fj = fi + 1 + t - (23 * fi - ((fi * (fi - 1)) >> 1));
    const unsigned pkr = (unsigned)(fi * D_) | ((unsigned)(fj * D_) << 16);
    const unsigned pkd = (unsigned)(F_ * D_) | ((unsigned)(F_ * D_) << 16);
    sdat[t] = make_float2(wv, __uint_as_float((t < I_) ? pkr : pkd));
  }
  __syncthreads();

  const int wid   = threadIdx.x >> 5;     // 0..7
  const int lane  = threadIdx.x & 31;
  const int half  = lane >> 4;
  const int sub   = lane & 15;
  const int ibase = wid * 16;

  // --- A fragment: 16 interaction rows x 32 gathered features (2 codewords).
  // K 0..15 = codebook[idx0], K 16..31 = codebook[idx1]; this lane takes the
  // half*8 slice of each -> two contiguous 32B (2x float4) gathered loads.
  int irow = ibase + sub; if (irow > I_ - 1) irow = I_ - 1;   // clamp padding
  const int* hp = hash_idx + ((size_t)b * I_ + irow) * 2;
  const int idx0 = hp[0];
  const int idx1 = hp[1];
  const float4* c0 = reinterpret_cast<const float4*>(codebook + (size_t)idx0 * 16 + half * 8);
  const float4* c1 = reinterpret_cast<const float4*>(codebook + (size_t)idx1 * 16 + half * 8);
  const float4 x0 = c0[0], x1 = c0[1], y0 = c1[0], y1 = c1[1];
  ABFrag a;
  a.v[0]  = tobf(x0.x); a.v[1]  = tobf(x0.y); a.v[2]  = tobf(x0.z); a.v[3]  = tobf(x0.w);
  a.v[4]  = tobf(x1.x); a.v[5]  = tobf(x1.y); a.v[6]  = tobf(x1.z); a.v[7]  = tobf(x1.w);
  a.v[8]  = tobf(y0.x); a.v[9]  = tobf(y0.y); a.v[10] = tobf(y0.z); a.v[11] = tobf(y0.w);
  a.v[12] = tobf(y1.x); a.v[13] = tobf(y1.y); a.v[14] = tobf(y1.z); a.v[15] = tobf(y1.w);

  // --- B fragment: W_t^T is [16 cols][32 K] bf16; lane takes K half*8.. and 16+half*8..
  ABFrag bf;
  bf.h[0] = *reinterpret_cast<const v8bf*>(WtT + sub * 32 + half * 8);
  bf.h[1] = *reinterpret_cast<const v8bf*>(WtT + sub * 32 + 16 + half * 8);
  const v16bf bm = bf.v;

  v8f acc = {};
  acc = __builtin_amdgcn_wmma_f32_16x16x32_bf16(
      false, a.v, false, bm, (short)0, acc, false, false);

  // --- scale by excitation weight and segment-sum into both endpoint fields
  const int ioff = ibase + half * 8;             // D rows: ioff + v
#pragma unroll
  for (int v = 0; v < 8; ++v) {
    const float2 dte = sdat[ioff + v];           // one ds_load_b64, no guard
    const float val = acc[v] * dte.x;            // 0.0 for padded rows
    const unsigned pk = __float_as_uint(dte.y);  // pad rows -> dummy slot
    atomicAdd(&sacc[(pk & 0xffffu) + sub], val); // ds_add_f32
    atomicAdd(&sacc[(pk >> 16) + sub], val);
  }
  __syncthreads();

  for (int t = threadIdx.x; t < F_ * D_; t += blockDim.x)
    out[(size_t)b * (F_ * D_) + t] = sacc[t];
}

// ---------------------------------------------------------------------------
extern "C" void kernel_launch(void* const* d_in, const int* in_sizes, int n_in,
                              void* d_out, int out_size, void* d_ws, size_t ws_size,
                              hipStream_t stream) {
  (void)in_sizes; (void)n_in; (void)out_size; (void)ws_size;
  const int*   hash_idx = (const int*)  d_in[0];   // [B, I, 2]
  const float* origin   = (const float*)d_in[1];   // [B, F, EO] == z [B, 384]
  const float* codebook = (const float*)d_in[2];   // [NB, 16]
  const float* Wt       = (const float*)d_in[3];   // [32, 16]
  const float* W1       = (const float*)d_in[4];   // [384, 384]
  const float* W2       = (const float*)d_in[5];   // [384, 123]
  float*       out      = (float*)d_out;           // [B, 24, 16]

  // Workspace layout (256B-aligned slabs)
  char* ws = (char*)d_ws;
  __bf16* zbf   = (__bf16*)(ws);                       // [4096,384]  3.0 MB
  __bf16* tmpbf = (__bf16*)(ws + 3145728);             // [4096,384]  3.0 MB
  __bf16* W1T   = (__bf16*)(ws + 6291456);             // [384,384]   288 KB
  __bf16* W2T   = (__bf16*)(ws + 6586368);             // [128,384]    96 KB
  __bf16* WtT   = (__bf16*)(ws + 6684672);             // [16,32]       1 KB
  float*  w     = (float*) (ws + 6685696);             // [4096,123]  2.0 MB

  // --- prep: all fp32->bf16 conversion / transposition hoisted here
  conv_bf16<<<(BATCH * ZDIM / 4 + 255) / 256, 256, 0, stream>>>(origin, zbf, BATCH * ZDIM / 4);
  transpose_bf16<<<(ZDIM * ZDIM + 255) / 256, 256, 0, stream>>>(W1, W1T, ZDIM, ZDIM, ZDIM);
  transpose_bf16<<<(NPAD2 * ZDIM + 255) / 256, 256, 0, stream>>>(W2, W2T, ZDIM, I_, NPAD2);
  transpose_bf16<<<2, 256, 0, stream>>>(Wt, WtT, 32, 16, 16);

  // GEMM1: tmp = z @ W1   (M=4096, N=384, K=384) -> 256 * 6 groups of 64 cols
  gemm_wmma_bf16<true><<<192, 256, 0, stream>>>(zbf, W1T, tmpbf, BATCH, ZDIM, ZDIM, 6);
  // GEMM2: w = tmp @ W2   (M=4096, N=123->128, K=384) -> 256 * 2 groups
  gemm_wmma_bf16<false><<<64, 256, 0, stream>>>(tmpbf, W2T, w, BATCH, I_, ZDIM, 2);
  // Fused gather + te WMMA + senetsum: one block per batch element
  fused_te_senetsum<<<BATCH, 256, 0, stream>>>(hash_idx, codebook, WtT, w, out);
}
